// GatConvNN_23828478558656
// MI455X (gfx1250) — matrix operand order repaired and checked by reading, not compiled, not run on hardware
//
#include <hip/hip_runtime.h>
#include <hip/hip_bf16.h>
#include <math.h>

typedef __attribute__((ext_vector_type(16))) _Float16 v16h;
typedef __attribute__((ext_vector_type(8)))  _Float16 v8h;
typedef __attribute__((ext_vector_type(8)))  float    v8f;

#define N_NODES 20000
#define N_PAIRS 200000
#define N_EDGES (4 * N_PAIRS)
#define NUM_GRAPHS 200
#define SORTN (1 << 20)          // padded sort size (>= N_EDGES)
#define KEYMASK 0xFFFFFull       // low 20 bits hold edge index (stable-sort tiebreak)

// ---------------------------------------------------------------------------
// CDNA5 WMMA fragment layouts (wave32):
// A (16x32 f16): lane L (row M=L%16), halves 0-7 -> K = kbase + 8*(L/16) + h
//                halves 8-15 -> K = kbase + 16 + 8*(L/16) + (h-8)
//   => two contiguous 8-element runs per lane per K-step.
// B (32x16 f16): lane L (col N=L%16), half h -> same K as A (table order)
// C/D (16x16 f32): lane L, reg r -> M = r + 8*(L/16), N = L%16
// ---------------------------------------------------------------------------
__device__ __forceinline__ int a_klocal(int lane, int h) {
  return h + ((h >= 8) ? 8 : 0) + 8 * (lane >> 4);
}
__device__ __forceinline__ int b_klocal(int lane, int h) {
  return h + ((h >= 8) ? 8 : 0) + 8 * (lane >> 4);
}

__device__ __forceinline__ v16h pack_a_f32(const float4 q0, const float4 q1,
                                           const float4 q2, const float4 q3) {
  v16h a;
  a[0] = (_Float16)q0.x;  a[1] = (_Float16)q0.y;
  a[2] = (_Float16)q0.z;  a[3] = (_Float16)q0.w;
  a[4] = (_Float16)q1.x;  a[5] = (_Float16)q1.y;
  a[6] = (_Float16)q1.z;  a[7] = (_Float16)q1.w;
  a[8] = (_Float16)q2.x;  a[9] = (_Float16)q2.y;
  a[10] = (_Float16)q2.z; a[11] = (_Float16)q2.w;
  a[12] = (_Float16)q3.x; a[13] = (_Float16)q3.y;
  a[14] = (_Float16)q3.z; a[15] = (_Float16)q3.w;
  return a;
}

__device__ __forceinline__ v16h cat8(v8h lo, v8h hi) {
  return __builtin_shufflevector(lo, hi, 0, 1, 2, 3, 4, 5, 6, 7,
                                 8, 9, 10, 11, 12, 13, 14, 15);
}

// generic pointer -> 32-bit LDS byte offset (for async-to-LDS destination)
__device__ __forceinline__ unsigned to_lds_off(const void* p) {
  return (unsigned)(unsigned long long)(__attribute__((address_space(3))) const void*)p;
}

__device__ __forceinline__ void atomicMaxFloat(float* addr, float v) {
  unsigned int* ua = (unsigned int*)addr;
  unsigned int old = *ua;
  while (true) {
    float f = __uint_as_float(old);
    if (f >= v) break;
    unsigned int assumed = old;
    old = atomicCAS(ua, assumed, __float_as_uint(v));
    if (old == assumed) break;
  }
}

// ---------------------------------------------------------------------------
// Small helper kernels
// ---------------------------------------------------------------------------
__global__ void fill_f32(float* p, float v, int n) {
  int i = blockIdx.x * blockDim.x + threadIdx.x;
  if (i < n) p[i] = v;
}

__global__ void f32_to_f16_k(const float* __restrict__ src, _Float16* __restrict__ dst,
                             int n) {
  int i = blockIdx.x * blockDim.x + threadIdx.x;
  if (i < n) dst[i] = (_Float16)src[i];
}

// c_table[cls][k] = tanh(emb_W[cls][k] + emb_b[k]); only 2 distinct rows exist
__global__ void c_table_k(const float* __restrict__ embW, const float* __restrict__ embb,
                          float* __restrict__ ct) {
  int t = threadIdx.x;
  if (t < 32) {
    int cls = t >> 4, k = t & 15;
    ct[t] = tanhf(embW[cls * 16 + k] + embb[k]);
  }
}

// deg + segment_sum(c, dst) using the 2-row c table
__global__ void edge_deg_k(const int* __restrict__ dst, const float* __restrict__ eattr,
                           const float* __restrict__ ct, float* __restrict__ deg,
                           float* __restrict__ loopc) {
  int e = blockIdx.x * blockDim.x + threadIdx.x;
  if (e >= N_EDGES) return;
  int d = dst[e];
  int cls = (int)eattr[e * 2 + 1];
  atomicAdd(&deg[d], 1.0f);
  for (int k = 0; k < 16; ++k) atomicAdd(&loopc[d * 16 + k], ct[cls * 16 + k]);
}

__global__ void loopc_div_k(float* __restrict__ loopc, const float* __restrict__ deg) {
  int i = blockIdx.x * blockDim.x + threadIdx.x;
  if (i < N_NODES * 16) loopc[i] /= fmaxf(deg[i / 16], 1.0f);
}

// wa[k] = sum_j We[k][j]*ae[j]; cetab[cls] = sum_k ct[cls][k]*wa[k]
__global__ void wa_k(const float* __restrict__ We, const float* __restrict__ ae,
                     const float* __restrict__ ct, float* __restrict__ wa,
                     float* __restrict__ cetab, int co) {
  int k = threadIdx.x;
  if (k < 16) {
    float s = 0.f;
    for (int j = 0; j < co; ++j) s += We[k * co + j] * ae[j];
    wa[k] = s;
  }
  __syncthreads();
  if (k < 2) {
    float s = 0.f;
    for (int q = 0; q < 16; ++q) s += ct[k * 16 + q] * wa[q];
    cetab[k] = s;
  }
}

__global__ void lce_k(const float* __restrict__ loopc, const float* __restrict__ wa,
                      float* __restrict__ lce) {
  int i = blockIdx.x * blockDim.x + threadIdx.x;
  if (i >= N_NODES) return;
  float s = 0.f;
  for (int k = 0; k < 16; ++k) s += loopc[i * 16 + k] * wa[k];
  lce[i] = s;
}

// ---------------------------------------------------------------------------
// Pack weight matrix [Kreal x Ncol] f32 into fragment-linear f16 B tiles:
// each lane's 16 halves per tile are contiguous (32 B) -> 2x b128 per fragment.
// ---------------------------------------------------------------------------
__global__ void pack_b_k(const float* __restrict__ W, _Float16* __restrict__ Bp,
                         int Kreal, int Kpad, int Ncol) {
  int t = blockIdx.x * blockDim.x + threadIdx.x;
  int total = (Kpad >> 5) * (Ncol >> 4) * 32 * 16;
  if (t >= total) return;
  int h = t & 15;
  int lane = (t >> 4) & 31;
  int tile = t >> 9;
  int ntiles = Ncol >> 4;
  int kt = tile / ntiles, nt = tile % ntiles;
  int K = kt * 32 + b_klocal(lane, h);
  int N = nt * 16 + (lane & 15);
  Bp[t] = (_Float16)((K < Kreal) ? W[K * Ncol + N] : 0.f);
}

// ---------------------------------------------------------------------------
// WMMA GEMM: xl[M x Ncol] = x[M x Kreal] @ W (packed Bp). One wave per 16x16
// output tile. VEC=true: unconditional float4 A loads (Kreal==Kpad, %8==0).
// VEC=false: branchless scalar A (clamped indices, select-to-zero).
// ---------------------------------------------------------------------------
template <bool VEC>
__global__ void gemm_xw_wmma(const float* __restrict__ x, const _Float16* __restrict__ Bp,
                             float* __restrict__ xl, int M, int Kpad, int Kreal, int Ncol) {
  int wave_global = (blockIdx.x * blockDim.x + threadIdx.x) >> 5;
  int lane = threadIdx.x & 31;
  int ntiles = Ncol >> 4;
  int mtiles = (M + 15) >> 4;
  if (wave_global >= ntiles * mtiles) return;  // wave-uniform exit keeps EXEC full
  int mt = wave_global / ntiles, nt = wave_global % ntiles;
  int mr = min(mt * 16 + (lane & 15), M - 1);  // clamp: OOB rows computed, not stored
  const float* xrow = x + (size_t)mr * Kreal;
  int g8 = (lane >> 4) * 8;
  v8f acc = {};
  for (int k0 = 0; k0 < Kpad; k0 += 32) {
    v16h a;
    if (VEC) {
      const float* p0 = xrow + k0 + g8;
      const float* p1 = xrow + k0 + 16 + g8;
      a = pack_a_f32(((const float4*)p0)[0], ((const float4*)p0)[1],
                     ((const float4*)p1)[0], ((const float4*)p1)[1]);
    } else {
#pragma unroll
      for (int h = 0; h < 16; ++h) {
        int ka = k0 + a_klocal(lane, h);
        float av = xrow[min(ka, Kreal - 1)];
        a[h] = (_Float16)((ka < Kreal) ? av : 0.f);
      }
    }
    const v16h b = *(const v16h*)(Bp + (((size_t)(k0 >> 5) * ntiles + nt) * 32 + lane) * 16);
    acc = __builtin_amdgcn_wmma_f32_16x16x32_f16(false, a, false, b, (short)0, acc,
                                                 false, false);
  }
  int mbase = mt * 16 + 8 * (lane >> 4);
  int ncol = nt * 16 + (lane & 15);
#pragma unroll
  for (int r = 0; r < 8; ++r) {
    int mm = mbase + r;
    if (mm < M) xl[(size_t)mm * Ncol + ncol] = acc[r];
  }
}

__global__ void al_k(const float* __restrict__ xl, const float* __restrict__ as_,
                     const float* __restrict__ ad_, float* __restrict__ als,
                     float* __restrict__ ald, int co) {
  int i = blockIdx.x * blockDim.x + threadIdx.x;
  if (i >= N_NODES) return;
  float s = 0.f, d = 0.f;
  for (int k = 0; k < co; ++k) {
    float v = xl[i * co + k];
    s += v * as_[k];
    d += v * ad_[k];
  }
  als[i] = s;
  ald[i] = d;
}

// alpha = leaky_relu(als[s]+ald[d]+ce), segment max into mbuf
__global__ void alpha_k(const int* __restrict__ src, const int* __restrict__ dst,
                        const float* __restrict__ eattr, const float* __restrict__ cetab,
                        const float* __restrict__ lce, const float* __restrict__ als,
                        const float* __restrict__ ald, float* __restrict__ alpha,
                        float* __restrict__ mbuf) {
  int idx = blockIdx.x * blockDim.x + threadIdx.x;
  if (idx >= N_EDGES + N_NODES) return;
  int s, d;
  float ce;
  if (idx < N_EDGES) {
    s = src[idx];
    d = dst[idx];
    ce = cetab[(int)eattr[idx * 2 + 1]];
  } else {
    s = d = idx - N_EDGES;
    ce = lce[s];
  }
  float a = als[s] + ald[d] + ce;
  a = (a > 0.f) ? a : 0.2f * a;
  alpha[idx] = a;
  atomicMaxFloat(&mbuf[d], a);
}

__global__ void exps_k(const int* __restrict__ dst, const float* __restrict__ alpha,
                       const float* __restrict__ mbuf, float* __restrict__ exb,
                       float* __restrict__ sbuf) {
  int idx = blockIdx.x * blockDim.x + threadIdx.x;
  if (idx >= N_EDGES + N_NODES) return;
  int d = (idx < N_EDGES) ? dst[idx] : (idx - N_EDGES);
  float ex = expf(alpha[idx] - mbuf[d]);
  exb[idx] = ex;
  atomicAdd(&sbuf[d], ex);
}

__global__ void agg_k(const int* __restrict__ src, const int* __restrict__ dst,
                      const float* __restrict__ exb, const float* __restrict__ sbuf,
                      const float* __restrict__ xl, float* __restrict__ agg, int co,
                      int total) {
  int t = blockIdx.x * blockDim.x + threadIdx.x;
  if (t >= total) return;
  int idx = t / co, k = t % co;
  int s, d;
  if (idx < N_EDGES) {
    s = src[idx];
    d = dst[idx];
  } else {
    s = d = idx - N_EDGES;
  }
  float w = exb[idx] / (sbuf[d] + 1e-16f);
  atomicAdd(&agg[d * co + k], w * xl[s * co + k]);
}

__global__ void bias_tanh_k(const float* __restrict__ agg, const float* __restrict__ b,
                            float* __restrict__ xo, int co) {
  int i = blockIdx.x * blockDim.x + threadIdx.x;
  if (i < N_NODES * co) xo[i] = tanhf(agg[i] + b[i % co]);
}

// ---------------------------------------------------------------------------
// Edge selection + stable sort keys: key = (src*N+dst)<<20 | edge_index
// ---------------------------------------------------------------------------
__global__ void build_keys_k(const int* __restrict__ src, const int* __restrict__ dst,
                             const unsigned char* __restrict__ det,
                             unsigned long long* __restrict__ keys, int* __restrict__ nkeep) {
  int e = blockIdx.x * blockDim.x + threadIdx.x;
  if (e >= SORTN) return;
  unsigned long long key = ~0ull;
  if (e < N_EDGES) {
    int s = src[e], d = dst[e];
    if (det[s] && det[d] && (s > d)) {
      key = (((unsigned long long)s * (unsigned long long)N_NODES +
              (unsigned long long)d) << 20) |
            (unsigned long long)e;
      atomicAdd(nkeep, 1);
    }
  }
  keys[e] = key;
}

__global__ void bitonic_k(unsigned long long* __restrict__ keys, int j, int k) {
  int i = blockIdx.x * blockDim.x + threadIdx.x;
  if (i >= SORTN) return;
  int ixj = i ^ j;
  if (ixj > i) {
    unsigned long long a = keys[i], b = keys[ixj];
    bool up = ((i & k) == 0);
    if ((a > b) == up) {
      keys[i] = b;
      keys[ixj] = a;
    }
  }
}

// ---------------------------------------------------------------------------
// Fused pair MLP: concat(x3h[s],x3h[d]) (256) -> 128 tanh -> 64 tanh -> 1
// One wave per 16-row tile. A: 2x b128 f16 gathers per fragment (branchless).
// d0 B: packed fragments from global (L2-resident). d1 B: async-staged into
// LDS at kernel entry (GLOBAL_LOAD_ASYNC_TO_LDS_B128, overlapped with the
// whole d0 phase, ASYNCcnt-synchronized), then read as ds_load_b128.
// ---------------------------------------------------------------------------
__global__ void __launch_bounds__(256) pair_mlp_k(
    const unsigned long long* __restrict__ keys, const int* __restrict__ nkeep_p,
    const int* __restrict__ src, const int* __restrict__ dst,
    const _Float16* __restrict__ x3h, const _Float16* __restrict__ d0p,
    const float* __restrict__ d0b, const _Float16* __restrict__ d1p,
    const float* __restrict__ d1b, const float* __restrict__ outw,
    const float* __restrict__ outb, float* __restrict__ vals) {
  __shared__ __align__(16) _Float16 h1[8][16 * 128];
  __shared__ __align__(16) _Float16 h2[8][16 * 64];
  __shared__ __align__(16) _Float16 d1s[4 * 4 * 512];  // 16 KB staged d1 weights
  const int lane = threadIdx.x & 31;
  const int wave = threadIdx.x >> 5;
  const int nkeep = *nkeep_p;
  const int row0 = (blockIdx.x * 8 + wave) * 16;
  const int g8 = (lane >> 4) * 8;

  // kick off async LDS staging of d1 weights: 256 threads x 4 iters x 16 B
  {
    unsigned dl = to_lds_off(&d1s[(size_t)threadIdx.x * 8]);
    const _Float16* gs = d1p + (size_t)threadIdx.x * 8;
#pragma unroll
    for (int it = 0; it < 4; ++it) {
      asm volatile("global_load_async_to_lds_b128 %0, %1, off"
                   :: "v"(dl + (unsigned)(it * 256 * 16)), "v"(gs + (size_t)it * 256 * 8)
                   : "memory");
    }
  }

  int r = row0 + (lane & 15);
  int ns = 0, nd = 0;
  if (r < nkeep) {  // inactive rows read node 0; their results are never stored
    int e = (int)(keys[r] & KEYMASK);
    ns = src[e];
    nd = dst[e];
  }
  const _Float16* rowA = x3h + (size_t)ns * 128;
  const _Float16* rowB = x3h + (size_t)nd * 128;

  // ---- layer d0: 16x256 @ 256x128 ----
  for (int nt = 0; nt < 8; ++nt) {
    v8f acc = {};
#pragma unroll
    for (int kk = 0; kk < 8; ++kk) {
      const int kbase = kk * 32;
      const int s0 = kbase + g8;        // halves 0-7: contiguous 8-run
      const int s1 = kbase + 16 + g8;   // halves 8-15: contiguous 8-run
      const _Float16* p0 = (s0 < 128) ? (rowA + s0) : (rowB + (s0 - 128));
      const _Float16* p1 = (s1 < 128) ? (rowA + s1) : (rowB + (s1 - 128));
      v16h a = cat8(*(const v8h*)p0, *(const v8h*)p1);
      const v16h b = *(const v16h*)(d0p + (((size_t)kk * 8 + nt) * 32 + lane) * 16);
      __builtin_prefetch(d0p + ((((size_t)kk + 1) * 8 + nt) * 32 + lane) * 16, 0, 1);
      acc = __builtin_amdgcn_wmma_f32_16x16x32_f16(false, a, false, b, (short)0, acc,
                                                   false, false);
    }
    int n = nt * 16 + (lane & 15);
    float bn = d0b[n];
#pragma unroll
    for (int rr = 0; rr < 8; ++rr) {
      int m = rr + 8 * (lane >> 4);
      h1[wave][m * 128 + n] = (_Float16)tanhf(acc[rr] + bn);
    }
  }
  // d1 weights must be resident in LDS before the next phase
  asm volatile("s_wait_asynccnt 0x0" ::: "memory");
  __syncthreads();

  // ---- layer d1: 16x128 @ 128x64 ----
  const _Float16* hrow = &h1[wave][(lane & 15) * 128];
  for (int nt = 0; nt < 4; ++nt) {
    v8f acc = {};
#pragma unroll
    for (int kk = 0; kk < 4; ++kk) {
      const int kbase = kk * 32;
      v16h a = cat8(*(const v8h*)(hrow + kbase + g8),
                    *(const v8h*)(hrow + kbase + 16 + g8));
      const v16h b = *(const v16h*)(d1s + (((size_t)kk * 4 + nt) * 32 + lane) * 16);
      acc = __builtin_amdgcn_wmma_f32_16x16x32_f16(false, a, false, b, (short)0, acc,
                                                   false, false);
    }
    int n = nt * 16 + (lane & 15);
    float bn = d1b[n];
#pragma unroll
    for (int rr = 0; rr < 8; ++rr) {
      int m = rr + 8 * (lane >> 4);
      h2[wave][m * 64 + n] = (_Float16)tanhf(acc[rr] + bn);
    }
  }
  __syncthreads();
  // ---- output head: 64 -> 1 ----
  if (lane < 16) {
    int rr2 = row0 + lane;
    if (rr2 < nkeep) {
      float s = outb[0];
      for (int k = 0; k < 64; ++k) s += (float)h2[wave][lane * 64 + k] * outw[k];
      vals[rr2] = s;
    }
  }
}

// ---------------------------------------------------------------------------
// Pairing, min/argmin, per-graph softmax
// ---------------------------------------------------------------------------
__global__ void pair_reduce_k(const unsigned long long* __restrict__ keys,
                              const int* __restrict__ nkeep_p, const int* __restrict__ src,
                              const int* __restrict__ dst, const float* __restrict__ eattr,
                              const int* __restrict__ batch, const float* __restrict__ vals,
                              float* __restrict__ pvals, int* __restrict__ peb,
                              float* __restrict__ gm, float* __restrict__ out,
                              int npair_out) {
  int p = blockIdx.x * blockDim.x + threadIdx.x;
  int np = min(npair_out, (*nkeep_p) >> 1);
  if (p >= np) return;
  float v0 = vals[2 * p], v1 = vals[2 * p + 1];
  int ind = (v1 < v0) ? 1 : 0;  // argmin, first index wins ties
  float v = ind ? v1 : v0;
  int e0 = (int)(keys[2 * p] & KEYMASK);
  int es = (int)(keys[2 * p + ind] & KEYMASK);
  int s = src[e0], d = dst[e0];
  int eb = batch[s];
  pvals[p] = v;
  peb[p] = eb;
  out[p] = (float)s;                           // e_out row 0
  out[npair_out + p] = (float)d;               // e_out row 1
  out[3 * npair_out + p] = eattr[es * 2 + 1];  // ec
  atomicMaxFloat(&gm[eb], v);
}

__global__ void pair_exp_k(const int* __restrict__ nkeep_p, const float* __restrict__ pvals,
                           const int* __restrict__ peb, const float* __restrict__ gm,
                           float* __restrict__ pex, float* __restrict__ gsum,
                           int npair_out) {
  int p = blockIdx.x * blockDim.x + threadIdx.x;
  int np = min(npair_out, (*nkeep_p) >> 1);
  if (p >= np) return;
  float ex = expf(pvals[p] - gm[peb[p]]);
  pex[p] = ex;
  atomicAdd(&gsum[peb[p]], ex);
}

__global__ void pair_out_k(const int* __restrict__ nkeep_p, const float* __restrict__ pex,
                           const int* __restrict__ peb, const float* __restrict__ gsum,
                           float* __restrict__ out, int npair_out) {
  int p = blockIdx.x * blockDim.x + threadIdx.x;
  int np = min(npair_out, (*nkeep_p) >> 1);
  if (p >= np) return;
  out[2 * npair_out + p] = pex[p] / (gsum[peb[p]] + 1e-16f);  // sm
}

// ---------------------------------------------------------------------------
// Host orchestration
// ---------------------------------------------------------------------------
extern "C" void kernel_launch(void* const* d_in, const int* in_sizes, int n_in,
                              void* d_out, int out_size, void* d_ws, size_t ws_size,
                              hipStream_t stream) {
  (void)in_sizes; (void)n_in; (void)ws_size;
  const float* x0 = (const float*)d_in[0];
  const int* edges = (const int*)d_in[1];
  const int* src = edges;
  const int* dst = edges + N_EDGES;
  const float* eattr = (const float*)d_in[2];
  const unsigned char* det = (const unsigned char*)d_in[3];
  const int* batch = (const int*)d_in[4];
  const float* embW = (const float*)d_in[5];
  const float* embb = (const float*)d_in[6];
  const float* d0W = (const float*)d_in[25];
  const float* d0b = (const float*)d_in[26];
  const float* d1W = (const float*)d_in[27];
  const float* d1b = (const float*)d_in[28];
  const float* outW = (const float*)d_in[29];
  const float* outb = (const float*)d_in[30];
  float* out = (float*)d_out;
  const int npair_out = out_size / 4;

  // --- carve workspace ---
  char* wp = (char*)d_ws;
  auto carve = [&](size_t bytes) {
    void* r = (void*)wp;
    wp += (bytes + 255) & ~(size_t)255;
    return r;
  };
  float* xbufA = (float*)carve((size_t)N_NODES * 128 * 4);
  float* xbufB = (float*)carve((size_t)N_NODES * 128 * 4);
  float* xl    = (float*)carve((size_t)N_NODES * 128 * 4);
  float* agg   = (float*)carve((size_t)N_NODES * 128 * 4);
  _Float16* x3h = (_Float16*)carve((size_t)N_NODES * 128 * 2);
  _Float16* Wp  = (_Float16*)carve((size_t)2 * 8 * 512 * 2);   // per-layer packed W
  _Float16* d0p = (_Float16*)carve((size_t)8 * 8 * 512 * 2);
  _Float16* d1p = (_Float16*)carve((size_t)4 * 4 * 512 * 2);
  float* ct    = (float*)carve(32 * 4);
  float* wa    = (float*)carve(16 * 4);
  float* cetab = (float*)carve(2 * 4);
  float* lce   = (float*)carve((size_t)N_NODES * 4);
  float* als   = (float*)carve((size_t)N_NODES * 4);
  float* ald   = (float*)carve((size_t)N_NODES * 4);
  float* deg   = (float*)carve((size_t)N_NODES * 4);
  float* loopc = (float*)carve((size_t)N_NODES * 16 * 4);
  float* alpha = (float*)carve((size_t)(N_EDGES + N_NODES) * 4);
  float* exb   = (float*)carve((size_t)(N_EDGES + N_NODES) * 4);
  float* mbuf  = (float*)carve((size_t)N_NODES * 4);
  float* sbuf  = (float*)carve((size_t)N_NODES * 4);
  unsigned long long* keys = (unsigned long long*)carve((size_t)SORTN * 8);
  int* nkeep   = (int*)carve(256);
  float* vals  = (float*)carve((size_t)N_EDGES * 4);
  float* pvals = (float*)carve((size_t)(N_EDGES / 2) * 4);
  float* pex   = (float*)carve((size_t)(N_EDGES / 2) * 4);
  int*   peb   = (int*)carve((size_t)(N_EDGES / 2) * 4);
  float* gm    = (float*)carve((size_t)NUM_GRAPHS * 4);
  float* gsum  = (float*)carve((size_t)NUM_GRAPHS * 4);

  const int B = 256;
  auto g = [](int n, int b) { return (n + b - 1) / b; };

  // edge embedding table + degree/loop_c (layer-invariant)
  c_table_k<<<1, 32, 0, stream>>>(embW, embb, ct);
  hipMemsetAsync(deg, 0, (size_t)N_NODES * 4, stream);
  hipMemsetAsync(loopc, 0, (size_t)N_NODES * 16 * 4, stream);
  edge_deg_k<<<g(N_EDGES, B), B, 0, stream>>>(dst, eattr, ct, deg, loopc);
  loopc_div_k<<<g(N_NODES * 16, B), B, 0, stream>>>(loopc, deg);

  // --- 3 GAT layers ---
  const int chans[4] = {5, 32, 64, 128};
  const float* xin = x0;
  float* bufs[2] = {xbufA, xbufB};
  for (int li = 0; li < 3; ++li) {
    int ci = chans[li], co = chans[li + 1];
    int Kpad = ((ci + 31) / 32) * 32;
    const float* W   = (const float*)d_in[7 + li * 6 + 0];
    const float* as_ = (const float*)d_in[7 + li * 6 + 1];
    const float* ad_ = (const float*)d_in[7 + li * 6 + 2];
    const float* We  = (const float*)d_in[7 + li * 6 + 3];
    const float* ae  = (const float*)d_in[7 + li * 6 + 4];
    const float* bb  = (const float*)d_in[7 + li * 6 + 5];

    pack_b_k<<<g((Kpad / 32) * (co / 16) * 512, B), B, 0, stream>>>(W, Wp, ci, Kpad, co);
    {
      int mtiles = (N_NODES + 15) / 16, ntiles = co / 16;
      int threads = mtiles * ntiles * 32;
      if ((ci == Kpad) && (ci % 8 == 0))
        gemm_xw_wmma<true><<<g(threads, B), B, 0, stream>>>(xin, Wp, xl, N_NODES, Kpad,
                                                            ci, co);
      else
        gemm_xw_wmma<false><<<g(threads, B), B, 0, stream>>>(xin, Wp, xl, N_NODES, Kpad,
                                                             ci, co);
    }
    al_k<<<g(N_NODES, B), B, 0, stream>>>(xl, as_, ad_, als, ald, co);
    wa_k<<<1, 32, 0, stream>>>(We, ae, ct, wa, cetab, co);
    lce_k<<<g(N_NODES, B), B, 0, stream>>>(loopc, wa, lce);
    fill_f32<<<g(N_NODES, B), B, 0, stream>>>(mbuf, -INFINITY, N_NODES);
    hipMemsetAsync(sbuf, 0, (size_t)N_NODES * 4, stream);
    alpha_k<<<g(N_EDGES + N_NODES, B), B, 0, stream>>>(src, dst, eattr, cetab, lce, als,
                                                       ald, alpha, mbuf);
    exps_k<<<g(N_EDGES + N_NODES, B), B, 0, stream>>>(dst, alpha, mbuf, exb, sbuf);
    hipMemsetAsync(agg, 0, (size_t)N_NODES * co * 4, stream);
    int total = (N_EDGES + N_NODES) * co;
    agg_k<<<g(total, B), B, 0, stream>>>(src, dst, exb, sbuf, xl, agg, co, total);
    float* xout = bufs[li & 1];
    bias_tanh_k<<<g(N_NODES * co, B), B, 0, stream>>>(agg, bb, xout, co);
    xin = xout;
  }
  // final node features -> f16 copy for the pair-MLP gathers
  f32_to_f16_k<<<g(N_NODES * 128, B), B, 0, stream>>>(xin, x3h, N_NODES * 128);

  // --- select kept edges + stable sort by (src*N+dst, edge_index) ---
  hipMemsetAsync(nkeep, 0, 4, stream);
  build_keys_k<<<g(SORTN, B), B, 0, stream>>>(src, dst, det, keys, nkeep);
  for (int k = 2; k <= SORTN; k <<= 1)
    for (int j = k >> 1; j > 0; j >>= 1)
      bitonic_k<<<SORTN / B, B, 0, stream>>>(keys, j, k);

  // --- fused pair MLP with WMMA + async LDS weight staging ---
  pack_b_k<<<g(8 * 8 * 512, B), B, 0, stream>>>(d0W, d0p, 256, 256, 128);
  pack_b_k<<<g(4 * 4 * 512, B), B, 0, stream>>>(d1W, d1p, 128, 128, 64);
  pair_mlp_k<<<g(N_EDGES, 128), 256, 0, stream>>>(keys, nkeep, src, dst, x3h, d0p, d0b,
                                                  d1p, d1b, outW, outb, vals);

  // --- pairing + per-graph softmax ---
  fill_f32<<<g(NUM_GRAPHS, B), B, 0, stream>>>(gm, -INFINITY, NUM_GRAPHS);
  hipMemsetAsync(gsum, 0, (size_t)NUM_GRAPHS * 4, stream);
  pair_reduce_k<<<g(N_EDGES / 2, B), B, 0, stream>>>(keys, nkeep, src, dst, eattr, batch,
                                                     vals, pvals, peb, gm, out, npair_out);
  pair_exp_k<<<g(N_EDGES / 2, B), B, 0, stream>>>(nkeep, pvals, peb, gm, pex, gsum,
                                                  npair_out);
  pair_out_k<<<g(N_EDGES / 2, B), B, 0, stream>>>(nkeep, pex, peb, gsum, out, npair_out);
}
